// SamplerQLSTM_65481071404996
// MI455X (gfx1250) — matrix operand order repaired and checked by reading, not compiled, not run on hardware
//
#include <hip/hip_runtime.h>
#include <hip/hip_bf16.h>
#include <stdint.h>

// Problem dims (fixed by reference)
#define T_STEPS 512
#define B_SZ    64
#define D_SZ    1024
#define H_SZ    1024
#define H2_SZ   2048

typedef __attribute__((ext_vector_type(16))) __bf16 v16bf;
typedef __attribute__((ext_vector_type(8)))  float  v8f;

// ---------- bf16 helpers (round-to-nearest-even) ----------
__device__ __forceinline__ uint16_t f2bf(float x) {
    union { float f; uint32_t u; } c; c.f = x;
    uint32_t u = c.u;
    uint32_t r = u + 0x7FFFu + ((u >> 16) & 1u);
    return (uint16_t)(r >> 16);
}
__device__ __forceinline__ float bf2f(uint16_t h) {
    union { uint32_t u; float f; } c; c.u = ((uint32_t)h) << 16;
    return c.f;
}
__device__ __forceinline__ uint32_t pack2(float a, float b) {
    return (uint32_t)f2bf(a) | ((uint32_t)f2bf(b) << 16);
}

// Fragment container: 8 dwords == 16 bf16 == one WMMA A or B operand (wave32)
union ABf { uint32_t u[8]; v16bf v; };

// ---------- one-time weight convert + transpose: W[K,N] f32 -> WT[N,K] bf16 ----------
__global__ __launch_bounds__(256) void k_transpose_bf16(
        const float* __restrict__ W, uint16_t* __restrict__ WT, int K, int N) {
    int idx = blockIdx.x * blockDim.x + threadIdx.x;
    if (idx >= K * N) return;
    int k = idx / N, n = idx % N;
    WT[(size_t)n * K + k] = f2bf(W[(size_t)k * N + n]);
}

// ---------- init hx_bf16 = 0, cx = 0 ----------
__global__ __launch_bounds__(256) void k_init(
        uint16_t* __restrict__ hx_bf, float* __restrict__ cx, int n) {
    int idx = blockIdx.x * blockDim.x + threadIdx.x;
    if (idx < n) { hx_bf[idx] = 0; cx[idx] = 0.0f; }
}

// =====================================================================
// Parallel pre-pass:  x_proj[T*B, H] = inputs[T*B, D] @ Wi + (bi + bh)
// Output fp32 into the `stacked` region of d_out (consumed at step t
// strictly before step t's update overwrites it with hx).
// Fragment layouts per CDNA5 ISA 7.12.2 (wave32):
//   A (16x32 bf16): lane m=lane&15, half=lane>>4; v[0..7]=K(half*8+0..7), v[8..15]=K(16+half*8+0..7)
//   B (32x16 bf16): lane n=lane&15, half=lane>>4; v[0..7]=B[K=half*8+0..7][n], v[8..15]=B[K=16+half*8+0..7][n] from WT[n][k]
//   C/D (16x16 f32): lane n=lane&15; acc[v] = C[m = v + 8*(lane>>4)][n]
// =====================================================================
__global__ __launch_bounds__(256) void k_gemm_xproj(
        const float*    __restrict__ x,    // [T*B, D] fp32
        const uint16_t* __restrict__ WiT,  // [H, D] bf16 (transposed)
        const float*    __restrict__ bi,
        const float*    __restrict__ bh,
        float*          __restrict__ xp)   // [T*B, H] fp32 out (stacked region)
{
    const int lane  = threadIdx.x & 31;
    const int wave  = blockIdx.x * (blockDim.x >> 5) + (threadIdx.x >> 5); // 131072 waves
    const int mtile = (wave >> 6) * 16;  // 2048 M-tiles (T*B = 32768)
    const int ntile = (wave & 63) * 16;  // 64 N-tiles   (H = 1024)
    const int half  = lane >> 4;
    const int mr    = lane & 15;
    const int nc    = lane & 15;

    v8f acc = {};
    for (int k0 = 0; k0 < D_SZ; k0 += 32) {
        ABf a, b;
        const float* ap = x + (size_t)(mtile + mr) * D_SZ + k0 + half * 8;
        float4 f0 = *(const float4*)(ap);
        float4 f1 = *(const float4*)(ap + 4);
        float4 f2 = *(const float4*)(ap + 16);
        float4 f3 = *(const float4*)(ap + 20);
        a.u[0] = pack2(f0.x, f0.y); a.u[1] = pack2(f0.z, f0.w);
        a.u[2] = pack2(f1.x, f1.y); a.u[3] = pack2(f1.z, f1.w);
        a.u[4] = pack2(f2.x, f2.y); a.u[5] = pack2(f2.z, f2.w);
        a.u[6] = pack2(f3.x, f3.y); a.u[7] = pack2(f3.z, f3.w);
        const uint16_t* bp = WiT + (size_t)(ntile + nc) * D_SZ + k0 + half * 8;
        uint4 q0 = *(const uint4*)(bp);
        uint4 q1 = *(const uint4*)(bp + 16);
        b.u[0] = q0.x; b.u[1] = q0.y; b.u[2] = q0.z; b.u[3] = q0.w;
        b.u[4] = q1.x; b.u[5] = q1.y; b.u[6] = q1.z; b.u[7] = q1.w;
        acc = __builtin_amdgcn_wmma_f32_16x16x32_bf16(false, a.v, false, b.v,
                                                      (short)0, acc, false, false);
    }

    const float bias = bi[ntile + nc] + bh[ntile + nc]; // fold both biases here
    #pragma unroll
    for (int v = 0; v < 8; ++v) {
        int m = mtile + v + half * 8;
        xp[(size_t)m * H_SZ + ntile + nc] = acc[v] + bias;
    }
}

// =====================================================================
// Per-step: combined = x_proj[t] + hx @ Wh   (bias already in x_proj)
// Pure bf16 load + wmma inner loop (K = 1024 -> 32 wmma / wave).
// =====================================================================
__global__ __launch_bounds__(256) void k_gemm_combined(
        const float*    __restrict__ xp_t,    // [B, H] fp32 (stacked + t*B*H)
        const uint16_t* __restrict__ hx_bf,   // [B, H] bf16
        const uint16_t* __restrict__ WhT,     // [H, H] bf16 (transposed)
        uint16_t*       __restrict__ comb_bf) // [B, H] bf16 out
{
    const int lane  = threadIdx.x & 31;
    const int wave  = blockIdx.x * (blockDim.x >> 5) + (threadIdx.x >> 5); // 256 waves
    const int mtile = (wave & 3) * 16;   // 4 M-tiles (B=64)
    const int ntile = (wave >> 2) * 16;  // 64 N-tiles (H=1024)
    const int half  = lane >> 4;
    const int mr    = lane & 15;
    const int nc    = lane & 15;

    v8f acc = {};
    for (int k0 = 0; k0 < H_SZ; k0 += 32) {
        ABf a, b;
        const uint16_t* ap = hx_bf + (size_t)(mtile + mr) * H_SZ + k0 + half * 8;
        uint4 p0 = *(const uint4*)(ap);
        uint4 p1 = *(const uint4*)(ap + 16);
        a.u[0] = p0.x; a.u[1] = p0.y; a.u[2] = p0.z; a.u[3] = p0.w;
        a.u[4] = p1.x; a.u[5] = p1.y; a.u[6] = p1.z; a.u[7] = p1.w;
        const uint16_t* bp = WhT + (size_t)(ntile + nc) * H_SZ + k0 + half * 8;
        uint4 q0 = *(const uint4*)(bp);
        uint4 q1 = *(const uint4*)(bp + 16);
        b.u[0] = q0.x; b.u[1] = q0.y; b.u[2] = q0.z; b.u[3] = q0.w;
        b.u[4] = q1.x; b.u[5] = q1.y; b.u[6] = q1.z; b.u[7] = q1.w;
        acc = __builtin_amdgcn_wmma_f32_16x16x32_bf16(false, a.v, false, b.v,
                                                      (short)0, acc, false, false);
    }

    #pragma unroll
    for (int v = 0; v < 8; ++v) {
        int m = mtile + v + half * 8;
        size_t off = (size_t)m * H_SZ + ntile + nc;
        comb_bf[off] = f2bf(acc[v] + xp_t[off]);
    }
}

// ---------- hidden_s = tanh(combined @ Ws1 + bs1) ----------
__global__ __launch_bounds__(256) void k_gemm_hidden(
        const uint16_t* __restrict__ comb_bf, // [B, H] bf16
        const uint16_t* __restrict__ Ws1T,    // [2H, H] bf16 (transposed)
        const float*    __restrict__ bs1,
        uint16_t*       __restrict__ hid_bf)  // [B, 2H] bf16 out
{
    const int lane  = threadIdx.x & 31;
    const int wave  = blockIdx.x * (blockDim.x >> 5) + (threadIdx.x >> 5); // 512 waves
    const int mtile = (wave & 3) * 16;   // 4 M-tiles
    const int ntile = (wave >> 2) * 16;  // 128 N-tiles (2H)
    const int half  = lane >> 4;
    const int mr    = lane & 15;
    const int nc    = lane & 15;

    v8f acc = {};
    for (int k0 = 0; k0 < H_SZ; k0 += 32) {
        ABf a, b;
        const uint16_t* ap = comb_bf + (size_t)(mtile + mr) * H_SZ + k0 + half * 8;
        uint4 p0 = *(const uint4*)(ap);
        uint4 p1 = *(const uint4*)(ap + 16);
        a.u[0] = p0.x; a.u[1] = p0.y; a.u[2] = p0.z; a.u[3] = p0.w;
        a.u[4] = p1.x; a.u[5] = p1.y; a.u[6] = p1.z; a.u[7] = p1.w;
        const uint16_t* bp = Ws1T + (size_t)(ntile + nc) * H_SZ + k0 + half * 8;
        uint4 q0 = *(const uint4*)(bp);
        uint4 q1 = *(const uint4*)(bp + 16);
        b.u[0] = q0.x; b.u[1] = q0.y; b.u[2] = q0.z; b.u[3] = q0.w;
        b.u[4] = q1.x; b.u[5] = q1.y; b.u[6] = q1.z; b.u[7] = q1.w;
        acc = __builtin_amdgcn_wmma_f32_16x16x32_bf16(false, a.v, false, b.v,
                                                      (short)0, acc, false, false);
    }

    const float bias = bs1[ntile + nc];
    #pragma unroll
    for (int v = 0; v < 8; ++v) {
        int m = mtile + v + half * 8;
        hid_bf[(size_t)m * H2_SZ + ntile + nc] = f2bf(tanhf(acc[v] + bias));
    }
}

// =====================================================================
// gate_probs = softmax(hidden_s @ Ws2 + bs2) : [64, 4]
// 1024 threads: (row, col) x 4 K-slices, dword-packed bf16 loads,
// LDS reduce, then per-(row,col) softmax.
// =====================================================================
__global__ __launch_bounds__(1024) void k_gates(
        const uint16_t* __restrict__ hid_bf, // [B, 2H]
        const float*    __restrict__ Ws2,    // [2H, 4]
        const float*    __restrict__ bs2,    // [4]
        float*          __restrict__ probs)  // [B, 4] out
{
    __shared__ float part[256][4];
    __shared__ float lg[B_SZ][4];

    const int tid   = threadIdx.x;
    const int rc    = tid >> 2;        // 0..255 : (row, col)
    const int slice = tid & 3;         // K-slice
    const int row   = rc >> 2;
    const int col   = rc & 3;

    const int kbeg = slice * (H2_SZ / 4);      // 512 K per slice
    const uint32_t* hp = (const uint32_t*)(hid_bf + (size_t)row * H2_SZ + kbeg);
    const float* wp = Ws2 + (size_t)kbeg * 4 + col;

    float acc = 0.0f;
    #pragma unroll 4
    for (int k = 0; k < H2_SZ / 8; ++k) {      // 256 iters, 2 bf16 per iter
        uint32_t hv = hp[k];
        acc += bf2f((uint16_t)(hv & 0xFFFFu)) * wp[8 * k];
        acc += bf2f((uint16_t)(hv >> 16))     * wp[8 * k + 4];
    }
    part[rc][slice] = acc;
    __syncthreads();

    if (tid < 256) {
        lg[tid >> 2][tid & 3] = part[tid][0] + part[tid][1] +
                                part[tid][2] + part[tid][3] + bs2[tid & 3];
    }
    __syncthreads();

    if (tid < 256) {
        int r = tid >> 2, c = tid & 3;
        float m = fmaxf(fmaxf(lg[r][0], lg[r][1]), fmaxf(lg[r][2], lg[r][3]));
        float s = expf(lg[r][0] - m) + expf(lg[r][1] - m) +
                  expf(lg[r][2] - m) + expf(lg[r][3] - m);
        probs[tid] = expf(lg[r][c] - m) / s;
    }
}

// ---------- cell update: cx = f*cx + i*tanh(g); hx = o*tanh(cx) ----------
__global__ __launch_bounds__(256) void k_update(
        const float* __restrict__ probs,     // [B, 4]
        float*       __restrict__ cx,        // [B, H] running (lives in d_out tail)
        float*       __restrict__ stacked_t, // d_out + t*B*H (overwrites x_proj[t])
        uint16_t*    __restrict__ hx_bf)     // [B, H] bf16 for next step
{
    int idx = blockIdx.x * blockDim.x + threadIdx.x; // B*H
    int b = idx >> 10; // H == 1024
    float f = probs[b * 4 + 0];
    float i = probs[b * 4 + 1];
    float g = probs[b * 4 + 2];
    float o = probs[b * 4 + 3];
    float c = f * cx[idx] + i * tanhf(g);
    cx[idx] = c;
    float h = o * tanhf(c);
    stacked_t[idx] = h;
    hx_bf[idx] = f2bf(h);
}

// ---------- final hx copy ----------
__global__ __launch_bounds__(256) void k_copy(
        const float* __restrict__ src, float* __restrict__ dst, int n) {
    int idx = blockIdx.x * blockDim.x + threadIdx.x;
    if (idx < n) dst[idx] = src[idx];
}

extern "C" void kernel_launch(void* const* d_in, const int* in_sizes, int n_in,
                              void* d_out, int out_size, void* d_ws, size_t ws_size,
                              hipStream_t stream) {
    const float* inputs = (const float*)d_in[0]; // [T, B, D]
    const float* Wi     = (const float*)d_in[1]; // [D, H]
    const float* bi     = (const float*)d_in[2]; // [H]
    const float* Wh     = (const float*)d_in[3]; // [H, H]
    const float* bh     = (const float*)d_in[4]; // [H]
    const float* Ws1    = (const float*)d_in[5]; // [H, 2H]
    const float* bs1    = (const float*)d_in[6]; // [2H]
    const float* Ws2    = (const float*)d_in[7]; // [2H, 4]
    const float* bs2    = (const float*)d_in[8]; // [4]

    float* out     = (float*)d_out;
    float* stacked = out;                                      // [T, B, H]
    float* hx_out  = out + (size_t)T_STEPS * B_SZ * H_SZ;      // [B, H]
    float* cx_out  = hx_out + (size_t)B_SZ * H_SZ;             // [B, H] (running cx)

    // Workspace layout (bf16 elements), ~8.5 MB total
    uint16_t* WiT     = (uint16_t*)d_ws;                       // H*D
    uint16_t* WhT     = WiT  + (size_t)H_SZ * D_SZ;            // H*H
    uint16_t* Ws1T    = WhT  + (size_t)H_SZ * H_SZ;            // 2H*H
    uint16_t* hx_bf   = Ws1T + (size_t)H2_SZ * H_SZ;           // B*H
    uint16_t* comb_bf = hx_bf + (size_t)B_SZ * H_SZ;           // B*H
    uint16_t* hid_bf  = comb_bf + (size_t)B_SZ * H_SZ;         // B*2H
    float*    probs   = (float*)(hid_bf + (size_t)B_SZ * H2_SZ); // B*4

    // One-time (per call) weight conversion + transpose into bf16 N-major form
    k_transpose_bf16<<<(D_SZ * H_SZ  + 255) / 256, 256, 0, stream>>>(Wi,  WiT,  D_SZ, H_SZ);
    k_transpose_bf16<<<(H_SZ * H_SZ  + 255) / 256, 256, 0, stream>>>(Wh,  WhT,  H_SZ, H_SZ);
    k_transpose_bf16<<<(H_SZ * H2_SZ + 255) / 256, 256, 0, stream>>>(Ws1, Ws1T, H_SZ, H2_SZ);
    k_init<<<(B_SZ * H_SZ + 255) / 256, 256, 0, stream>>>(hx_bf, cx_out, B_SZ * H_SZ);

    // Fully parallel input-projection WMMA pre-pass (no serial dependence):
    // x_proj (fp32, biases folded) lands in the stacked region of d_out.
    // 131072 waves = 16384 blocks of 8 waves.
    k_gemm_xproj<<<16384, 256, 0, stream>>>(inputs, WiT, bi, bh, stacked);

    for (int t = 0; t < T_STEPS; ++t) {
        k_gemm_combined<<<32, 256, 0, stream>>>(
            stacked + (size_t)t * B_SZ * H_SZ, hx_bf, WhT, comb_bf);
        k_gemm_hidden<<<64, 256, 0, stream>>>(comb_bf, Ws1T, bs1, hid_bf);
        k_gates<<<1, 1024, 0, stream>>>(hid_bf, Ws2, bs2, probs);
        k_update<<<(B_SZ * H_SZ) / 256, 256, 0, stream>>>(
            probs, cx_out, stacked + (size_t)t * B_SZ * H_SZ, hx_bf);
    }
    k_copy<<<(B_SZ * H_SZ + 255) / 256, 256, 0, stream>>>(
        stacked + (size_t)(T_STEPS - 1) * B_SZ * H_SZ, hx_out, B_SZ * H_SZ);
}
